// SelfAttention_1537598292560
// MI455X (gfx1250) — compile-verified, matrix operand order checked by reading, not run on hardware
//
#include <hip/hip_runtime.h>
#include <hip/hip_bf16.h>

// ---------------------------------------------------------------------------
// Types for CDNA5 WMMA (wave32): D(f32 16x16) = A(bf16 16x32) * B(bf16 32x16) + C
// ---------------------------------------------------------------------------
typedef __attribute__((ext_vector_type(16))) __bf16 v16bf;
typedef __attribute__((ext_vector_type(8)))  float  v8f;

struct FragBF {
    union { v16bf v; uint4 q[2]; };
};

__device__ inline v8f wmma_bf16(const FragBF& a, const FragBF& b, v8f c) {
    return __builtin_amdgcn_wmma_f32_16x16x32_bf16(
        /*neg_a=*/false, a.v, /*neg_b=*/false, b.v,
        /*c_mod=*/(short)0, c, /*reuse_a=*/false, /*reuse_b=*/false);
}

__device__ inline unsigned short f2bf(float f) {
    unsigned int u = __float_as_uint(f);
    unsigned int r = (u + 0x7FFFu + ((u >> 16) & 1u)) >> 16;
    return (unsigned short)r;
}

// CDNA5 async copy: per-lane 16B global -> LDS, tracked by ASYNCcnt.
__device__ inline void async_ld16(void* lds, const void* gaddr) {
    asm volatile("global_load_async_to_lds_b128 %0, %1, off"
                 :: "v"((unsigned)(size_t)lds), "v"(gaddr)
                 : "memory");
}
__device__ inline void wait_async_le4() {
    asm volatile("s_wait_asynccnt 0x4" ::: "memory");
}
__device__ inline void wait_async_le8() {
    asm volatile("s_wait_asynccnt 0x8" ::: "memory");
}
__device__ inline void wait_async_0() {
    asm volatile("s_wait_asynccnt 0x0" ::: "memory");
}

// ---------------------------------------------------------------------------
// Constants (fixed by setup_inputs)
// ---------------------------------------------------------------------------
#define BATCH    8
#define SEQ      512
#define DIM      4096
#define NHEADS   32
#define NKVH     8
#define HDIM     128
#define MAXSEQ   2048
#define MROWS    (BATCH * SEQ)   // 4096

// ---------------------------------------------------------------------------
// 1) f32 -> bf16 convert
// ---------------------------------------------------------------------------
__global__ __launch_bounds__(256) void cvt_f32_bf16(
    const float* __restrict__ in, unsigned short* __restrict__ out, int n) {
    int i = blockIdx.x * 256 + threadIdx.x;
    if (i < n) out[i] = f2bf(in[i]);
}

// ---------------------------------------------------------------------------
// 2) GEMM: C[M,N] = A[M,K] * W[N,K]^T, bf16 in, f32 out.
//    Block tile 128x128, K-step 32, double-buffered LDS filled by
//    global_load_async_to_lds_b128. 8 waves (4x2), each 32x64 (2x4 WMMA accs).
// ---------------------------------------------------------------------------
__global__ __launch_bounds__(256) void gemm_bf16_nt(
    const unsigned short* __restrict__ A,
    const unsigned short* __restrict__ W,
    float* __restrict__ C, int M, int N, int K) {
    __shared__ unsigned short sm[2][8192];   // per buf: A 128x32 + B 128x32

    const int tid  = threadIdx.x;
    const int wave = tid >> 5;
    const int lane = tid & 31;
    const int wm = wave >> 1;
    const int wn = wave & 1;
    const int m0 = blockIdx.y * 128;
    const int n0 = blockIdx.x * 128;
    const int qrow  = lane & 15;
    const int khalf = lane >> 4;

    const int rA0 = tid >> 2, part = tid & 3;
    const unsigned short* gA0 = A + (size_t)(m0 + rA0)      * K + part * 8;
    const unsigned short* gA1 = A + (size_t)(m0 + 64 + rA0) * K + part * 8;
    const unsigned short* gB0 = W + (size_t)(n0 + rA0)      * K + part * 8;
    const unsigned short* gB1 = W + (size_t)(n0 + 64 + rA0) * K + part * 8;
    unsigned short* dA0 = &sm[0][tid * 8];
    unsigned short* dA1 = &sm[0][(tid + 256) * 8];
    unsigned short* dB0 = &sm[0][4096 + tid * 8];
    unsigned short* dB1 = &sm[0][4096 + (tid + 256) * 8];

    v8f acc[2][4] = {};
    const int nk = K >> 5;

    async_ld16(dA0, gA0);
    async_ld16(dA1, gA1);
    async_ld16(dB0, gB0);
    async_ld16(dB1, gB1);

    for (int i = 0; i < nk; ++i) {
        const int cur = i & 1;
        if (i + 1 < nk) {
            const int k1 = (i + 1) << 5;
            const size_t boff = (cur ^ 1) ? 8192u : 0u;
            async_ld16(dA0 + boff, gA0 + k1);
            async_ld16(dA1 + boff, gA1 + k1);
            async_ld16(dB0 + boff, gB0 + k1);
            async_ld16(dB1 + boff, gB1 + k1);
            wait_async_le4();
        } else {
            wait_async_0();
        }
        __syncthreads();

        const unsigned short* bufA = sm[cur];
        const unsigned short* bufB = sm[cur] + 4096;

        FragBF af[2], bf[4];
        #pragma unroll
        for (int mi = 0; mi < 2; ++mi) {
            int row = wm * 32 + mi * 16 + qrow;
            const uint4* p = reinterpret_cast<const uint4*>(bufA + row * 32 + khalf * 8);
            af[mi].q[0] = p[0];
            af[mi].q[1] = p[2];
        }
        #pragma unroll
        for (int ni = 0; ni < 4; ++ni) {
            int col = wn * 64 + ni * 16 + qrow;
            const uint4* p = reinterpret_cast<const uint4*>(bufB + col * 32 + khalf * 8);
            bf[ni].q[0] = p[0];
            bf[ni].q[1] = p[2];
        }
        #pragma unroll
        for (int mi = 0; mi < 2; ++mi)
            #pragma unroll
            for (int ni = 0; ni < 4; ++ni)
                acc[mi][ni] = wmma_bf16(af[mi], bf[ni], acc[mi][ni]);

        __syncthreads();
    }

    #pragma unroll
    for (int mi = 0; mi < 2; ++mi)
        #pragma unroll
        for (int ni = 0; ni < 4; ++ni) {
            int col   = n0 + wn * 64 + ni * 16 + qrow;
            int rbase = m0 + wm * 32 + mi * 16 + khalf * 8;
            #pragma unroll
            for (int i = 0; i < 8; ++i)
                C[(size_t)(rbase + i) * N + col] = acc[mi][ni][i];
        }
}

// ---------------------------------------------------------------------------
// 3) RoPE on xq -> bf16 q[b][h][s][d]
// ---------------------------------------------------------------------------
__global__ __launch_bounds__(256) void rope_pack_q(
    const float* __restrict__ xqf, unsigned short* __restrict__ qbf,
    const int* __restrict__ sp) {
    int idx = blockIdx.x * 256 + threadIdx.x;   // (b, s, h, dp)
    int dp = idx & 63;  idx >>= 6;
    int h  = idx & 31;  idx >>= 5;
    int s  = idx & 511;
    int b  = idx >> 9;
    int start = sp[0];

    size_t ri = ((size_t)(b * SEQ + s)) * DIM + h * HDIM + 2 * dp;
    float x0 = xqf[ri], x1 = xqf[ri + 1];
    float freq = (float)(start + s) * __powf(10000.f, -(float)(2 * dp) / 128.f);
    float sn, cs;
    __sincosf(freq, &sn, &cs);
    float o0 = x0 * cs - x1 * sn;
    float o1 = x0 * sn + x1 * cs;

    size_t oi = (((size_t)(b * NHEADS + h) * SEQ + s)) * HDIM + 2 * dp;
    qbf[oi]     = f2bf(o0);
    qbf[oi + 1] = f2bf(o1);
}

// ---------------------------------------------------------------------------
// 4) Build keys (bf16, [b][kvh][k][d]) and V^T (bf16, [b][kvh][d][k])
// ---------------------------------------------------------------------------
__global__ __launch_bounds__(256) void build_kv(
    const float* __restrict__ cache_k, const float* __restrict__ cache_v,
    const float* __restrict__ xkf, const float* __restrict__ xvf,
    const int* __restrict__ sp,
    unsigned short* __restrict__ keys, unsigned short* __restrict__ vT) {
    int idx = blockIdx.x * 256 + threadIdx.x;   // (b, k, h, dp)
    int dp = idx & 63;   idx >>= 6;
    int h  = idx & 7;    idx >>= 3;
    int k  = idx & 2047;
    int b  = idx >> 11;
    int start = sp[0];
    if (k >= start + SEQ) return;

    float k0, k1, v0, v1;
    if (k < start) {
        size_t ci = (((size_t)(b * MAXSEQ + k) * NKVH + h)) * HDIM + 2 * dp;
        k0 = cache_k[ci]; k1 = cache_k[ci + 1];
        v0 = cache_v[ci]; v1 = cache_v[ci + 1];
    } else {
        int s = k - start;
        size_t ri = ((size_t)(b * SEQ + s)) * (NKVH * HDIM) + h * HDIM + 2 * dp;
        float a0 = xkf[ri], a1 = xkf[ri + 1];
        float freq = (float)k * __powf(10000.f, -(float)(2 * dp) / 128.f);
        float sn, cs;
        __sincosf(freq, &sn, &cs);
        k0 = a0 * cs - a1 * sn;
        k1 = a0 * sn + a1 * cs;
        v0 = xvf[ri]; v1 = xvf[ri + 1];
    }
    size_t ki = (((size_t)(b * NKVH + h) * MAXSEQ + k)) * HDIM + 2 * dp;
    keys[ki]     = f2bf(k0);
    keys[ki + 1] = f2bf(k1);
    size_t vi = ((size_t)(b * NKVH + h) * HDIM + 2 * dp) * MAXSEQ + k;
    vT[vi]          = f2bf(v0);
    vT[vi + MAXSEQ] = f2bf(v1);
}

// ---------------------------------------------------------------------------
// 5) Flash attention: block = (b, kvh, q-tile16); 4 waves = the 4 Q-heads
//    sharing this KV head. K/V chunks are async-copied once per block into
//    double-buffered LDS and consumed by all 4 waves (4x traffic reduction).
// ---------------------------------------------------------------------------
__global__ __launch_bounds__(128) void flash_attn(
    const unsigned short* __restrict__ qbf,
    const unsigned short* __restrict__ keys,
    const unsigned short* __restrict__ vT,
    unsigned short* __restrict__ ctx,
    const int* __restrict__ sp) {
    __shared__ unsigned short Kbuf[2][32 * 128];   // [key][d]   8KB each
    __shared__ unsigned short Vbuf[2][128 * 32];   // [d][key]   8KB each
    __shared__ unsigned short P[4][16 * 32];       // per-wave P transpose

    const int tid  = threadIdx.x;
    const int wave = tid >> 5;
    const int lane = tid & 31;
    const int blk  = blockIdx.x;          // (b, kvh, qt): 8*8*32
    const int qt   = blk & 31;
    const int kvh  = (blk >> 5) & 7;
    const int b    = blk >> 8;
    const int h    = kvh * 4 + wave;      // each wave = one Q head of this group
    const int qb   = qt * 16;
    const int qrow  = lane & 15;
    const int khalf = lane >> 4;
    const int kvlen = sp[0] + SEQ;        // 2048

    const unsigned short* Qp = qbf  + (((size_t)(b * NHEADS + h) * SEQ) + qb) * HDIM;
    const unsigned short* Kp = keys + ((size_t)(b * NKVH + kvh) * MAXSEQ) * HDIM;
    const unsigned short* Vp = vT   + ((size_t)(b * NKVH + kvh) * HDIM) * MAXSEQ;

    // Q fragments (per wave/head): 16x128 as 4 K-chunks of 32
    FragBF qf[4];
    #pragma unroll
    for (int kk = 0; kk < 4; ++kk) {
        const uint4* p = reinterpret_cast<const uint4*>(
            Qp + (size_t)qrow * HDIM + kk * 32 + khalf * 8);
        qf[kk].q[0] = p[0];
        qf[kk].q[1] = p[2];
    }

    // async K/V chunk copy: 512 + 512 16B segments, 8 per thread
    auto issue = [&](int buf, int kc) {
        #pragma unroll
        for (int j = 0; j < 4; ++j) {
            int s = tid + j * 128;
            int row = s >> 4, prt = s & 15;            // 16 segs per 128-d key row
            async_ld16(&Kbuf[buf][s * 8], Kp + (size_t)(kc + row) * HDIM + prt * 8);
        }
        #pragma unroll
        for (int j = 0; j < 4; ++j) {
            int s = tid + j * 128;
            int d = s >> 2, prt = s & 3;               // 4 segs per 32-key d row
            async_ld16(&Vbuf[buf][s * 8], Vp + (size_t)d * MAXSEQ + kc + prt * 8);
        }
    };

    float mrow[8], lrow[8];
    #pragma unroll
    for (int i = 0; i < 8; ++i) { mrow[i] = -1e30f; lrow[i] = 0.f; }
    v8f o[8] = {};
    const float scale = 0.08838834764831845f;   // 1/sqrt(128)
    unsigned short* myP = P[wave];

    const int nchunk = kvlen >> 5;
    issue(0, 0);

    for (int ic = 0; ic < nchunk; ++ic) {
        const int cur = ic & 1;
        if (ic + 1 < nchunk) {
            issue(cur ^ 1, (ic + 1) << 5);
            wait_async_le8();           // chunk ic landed; ic+1 in flight
        } else {
            wait_async_0();
        }
        __syncthreads();

        const unsigned short* kb = Kbuf[cur];
        const unsigned short* vb = Vbuf[cur];

        // ---- scores: two 16x16 tiles covering 32 keys ----
        v8f s0 = {}, s1 = {};
        #pragma unroll
        for (int kk = 0; kk < 4; ++kk) {
            FragBF bk0, bk1;
            const uint4* p0 = reinterpret_cast<const uint4*>(
                kb + (size_t)qrow * HDIM + kk * 32 + khalf * 8);
            bk0.q[0] = p0[0]; bk0.q[1] = p0[2];
            const uint4* p1 = reinterpret_cast<const uint4*>(
                kb + (size_t)(16 + qrow) * HDIM + kk * 32 + khalf * 8);
            bk1.q[0] = p1[0]; bk1.q[1] = p1[2];
            s0 = wmma_bf16(qf[kk], bk0, s0);
            s1 = wmma_bf16(qf[kk], bk1, s1);
        }

        // ---- online softmax ----
        float alpha[8];
        #pragma unroll
        for (int i = 0; i < 8; ++i) {
            float sm0 = s0[i] * scale;
            float sm1 = s1[i] * scale;
            float cm = fmaxf(sm0, sm1);
            #pragma unroll
            for (int d = 1; d < 16; d <<= 1) cm = fmaxf(cm, __shfl_xor(cm, d, 32));
            float mn = fmaxf(mrow[i], cm);
            float a  = __expf(mrow[i] - mn);
            float p0 = __expf(sm0 - mn);
            float p1 = __expf(sm1 - mn);
            float rs = p0 + p1;
            #pragma unroll
            for (int d = 1; d < 16; d <<= 1) rs += __shfl_xor(rs, d, 32);
            lrow[i] = lrow[i] * a + rs;
            mrow[i] = mn;
            alpha[i] = a;
            int prow = i + khalf * 8;
            myP[prow * 32 + qrow]      = f2bf(p0);
            myP[prow * 32 + 16 + qrow] = f2bf(p1);
        }
        #pragma unroll
        for (int nt = 0; nt < 8; ++nt)
            #pragma unroll
            for (int i = 0; i < 8; ++i) o[nt][i] *= alpha[i];

        // ---- P as A-fragment via per-wave LDS transpose ----
        FragBF pf;
        {
            const uint4* p = reinterpret_cast<const uint4*>(myP + qrow * 32 + khalf * 8);
            pf.q[0] = p[0];
            pf.q[1] = p[2];
        }

        // ---- context: O[16x128] += P[16x32] * V[32x128] ----
        #pragma unroll
        for (int nt = 0; nt < 8; ++nt) {
            FragBF bv;
            const uint4* p = reinterpret_cast<const uint4*>(
                vb + (size_t)(nt * 16 + qrow) * 32 + khalf * 8);
            bv.q[0] = p[0]; bv.q[1] = p[2];
            o[nt] = wmma_bf16(pf, bv, o[nt]);
        }

        __syncthreads();   // all waves done with buffer `cur`
    }

    #pragma unroll
    for (int nt = 0; nt < 8; ++nt)
        #pragma unroll
        for (int i = 0; i < 8; ++i) {
            float v = o[nt][i] / lrow[i];
            int row = qb + i + khalf * 8;
            size_t oi = ((size_t)(b * SEQ) + row) * DIM + h * HDIM + nt * 16 + qrow;
            ctx[oi] = f2bf(v);
        }
}

// ---------------------------------------------------------------------------
// Host-side launch
// ---------------------------------------------------------------------------
extern "C" void kernel_launch(void* const* d_in, const int* in_sizes, int n_in,
                              void* d_out, int out_size, void* d_ws, size_t ws_size,
                              hipStream_t stream) {
    const float* x       = (const float*)d_in[0];
    const float* Wq      = (const float*)d_in[1];
    const float* Wk      = (const float*)d_in[2];
    const float* Wv      = (const float*)d_in[3];
    const float* Wo      = (const float*)d_in[4];
    const float* cache_k = (const float*)d_in[5];
    const float* cache_v = (const float*)d_in[6];
    const int*   sp      = (const int*)d_in[7];

    unsigned char* ws = (unsigned char*)d_ws;
    size_t off = 0;
    auto alloc = [&](size_t bytes) {
        void* p = ws + off;
        off += (bytes + 255) & ~(size_t)255;
        return p;
    };

    const size_t NX  = (size_t)MROWS * DIM;            // 16.7M
    const size_t NKV = (size_t)MROWS * (NKVH * HDIM);  // 4.2M

    unsigned short* xbf  = (unsigned short*)alloc(NX * 2);
    unsigned short* wqb  = (unsigned short*)alloc((size_t)DIM * DIM * 2);
    unsigned short* wkb  = (unsigned short*)alloc((size_t)NKVH * HDIM * DIM * 2);
    unsigned short* wvb  = (unsigned short*)alloc((size_t)NKVH * HDIM * DIM * 2);
    unsigned short* wob  = (unsigned short*)alloc((size_t)DIM * DIM * 2);
    float*          xqf  = (float*)alloc(NX * 4);
    float*          xkf  = (float*)alloc(NKV * 4);
    float*          xvf  = (float*)alloc(NKV * 4);
    unsigned short* qbf  = (unsigned short*)alloc(NX * 2);
    unsigned short* keys = (unsigned short*)alloc((size_t)BATCH * NKVH * MAXSEQ * HDIM * 2);
    unsigned short* vT   = (unsigned short*)alloc((size_t)BATCH * NKVH * MAXSEQ * HDIM * 2);
    unsigned short* ctx  = (unsigned short*)alloc(NX * 2);

    cvt_f32_bf16<<<(int)(NX / 256), 256, 0, stream>>>(x,  xbf, (int)NX);
    cvt_f32_bf16<<<(int)(((size_t)DIM * DIM) / 256), 256, 0, stream>>>(Wq, wqb, DIM * DIM);
    cvt_f32_bf16<<<(int)(((size_t)NKVH * HDIM * DIM) / 256), 256, 0, stream>>>(Wk, wkb, NKVH * HDIM * DIM);
    cvt_f32_bf16<<<(int)(((size_t)NKVH * HDIM * DIM) / 256), 256, 0, stream>>>(Wv, wvb, NKVH * HDIM * DIM);
    cvt_f32_bf16<<<(int)(((size_t)DIM * DIM) / 256), 256, 0, stream>>>(Wo, wob, DIM * DIM);

    gemm_bf16_nt<<<dim3(DIM / 128, MROWS / 128), 256, 0, stream>>>(xbf, wqb, xqf, MROWS, DIM, DIM);
    gemm_bf16_nt<<<dim3((NKVH * HDIM) / 128, MROWS / 128), 256, 0, stream>>>(xbf, wkb, xkf, MROWS, NKVH * HDIM, DIM);
    gemm_bf16_nt<<<dim3((NKVH * HDIM) / 128, MROWS / 128), 256, 0, stream>>>(xbf, wvb, xvf, MROWS, NKVH * HDIM, DIM);

    rope_pack_q<<<(BATCH * SEQ * NHEADS * 64) / 256, 256, 0, stream>>>(xqf, qbf, sp);

    build_kv<<<(BATCH * MAXSEQ * NKVH * 64) / 256, 256, 0, stream>>>(
        cache_k, cache_v, xkf, xvf, sp, keys, vT);

    flash_attn<<<BATCH * NKVH * (SEQ / 16), 128, 0, stream>>>(qbf, keys, vT, ctx, sp);

    gemm_bf16_nt<<<dim3(DIM / 128, MROWS / 128), 256, 0, stream>>>(
        ctx, wob, (float*)d_out, MROWS, DIM, DIM);

    (void)in_sizes; (void)n_in; (void)out_size; (void)ws_size;
}
